// DLRM_Net_Habana_74904229642568
// MI455X (gfx1250) — matrix operand order repaired and checked by reading, not compiled, not run on hardware
//
#include <hip/hip_runtime.h>

// ---------------------------------------------------------------------------
// DLRM forward for MI455X (gfx1250, wave32, WMMA + async-to-LDS staging).
// bf16 everywhere: weights AND the fp32 dense input are pre-converted and
// zero-padded once, activations are stored bf16, so every GEMM stage is a
// pure b128 copy -- done with GLOBAL_LOAD_ASYNC_TO_LDS_B128 (ASYNCcnt),
// feeding the v_wmma_f32_16x16x32_bf16 core.
// ---------------------------------------------------------------------------

typedef __attribute__((ext_vector_type(16))) __bf16 v16bf;
typedef __attribute__((ext_vector_type(8)))  float  v8f;

#define B_SZ     4096
#define NTAB     26
#define VOCAB_SZ 100000
#define EMB_D    64
#define NNZ_SZ   8192
#define TROWS    27                 // 1 + NTAB
#define TSTRIDE  (TROWS * EMB_D)    // 1728 floats per batch row of T
#define RCOLS    415                // 64 + 351
#define RPAD     416                // padded row stride of R (bf16)
#define NTRI     351                // strict lower-triangle count for 27x27

#define BM 128
#define BN 64
#define BK 32

// fp32 -> bf16, round-half-up (single v_add + shift / d16_hi store)
__device__ __forceinline__ unsigned short f2bf(float f) {
  return (unsigned short)((__float_as_uint(f) + 0x8000u) >> 16);
}
// pack two fp32 -> bf16x2 dword: v_add, v_add, v_perm_b32
__device__ __forceinline__ unsigned int pk2(float a, float b) {
  const unsigned int ua = __float_as_uint(a) + 0x8000u;
  const unsigned int ub = __float_as_uint(b) + 0x8000u;
  return __builtin_amdgcn_perm(ub, ua, 0x07060302u);
}

struct FragBF { union { v16bf v; uint4 q[2]; }; };

// LDS-relative byte address (addrspace(3) offset) of a shared-memory pointer.
__device__ __forceinline__ unsigned lds_addr(const void* p) {
  return (unsigned)(uintptr_t)(__attribute__((address_space(3))) const void*)p;
}
// Async DMA: 16 bytes per lane, global -> LDS, tracked by ASYNCcnt.
__device__ __forceinline__ void async_ld_b128(unsigned lds, const void* gp) {
  asm volatile("global_load_async_to_lds_b128 %0, %1, off"
               :: "v"(lds), "v"(gp) : "memory");
}
__device__ __forceinline__ void wait_async0() {
  asm volatile("s_wait_asynccnt 0" ::: "memory");
}

// ---------------------------------------------------------------------------
// fp32 -> bf16 row convert with zero padding: out[n*Kp+k] = bf16(W[n*K+k])
// for k<K else 0. Grid: (ceil(Kp/256), Nrows). Used for all weights AND the
// dense input matrix.
// ---------------------------------------------------------------------------
__global__ void __launch_bounds__(256) convert_pad_rows(
    const float* __restrict__ W, unsigned short* __restrict__ out,
    int K, int Kp)
{
  const int k = blockIdx.x * 256 + threadIdx.x;
  const int n = blockIdx.y;
  if (k >= Kp) return;
  out[(size_t)n * Kp + k] = (k < K) ? f2bf(W[(size_t)n * K + k]) : (unsigned short)0;
}

// ---------------------------------------------------------------------------
// GEMM: C[M,N] = act(A[M,K] * W[N,K]^T + bias[N]),  act 0=relu 1=sigmoid.
// A, W: bf16, zero-padded, row strides lda / Kp (Kp multiple of 32).
// C: bf16 (row stride ldc) unless cF32. Block tile 128x64, K-step 32,
// 8 waves each owning a 16(M) x 64(N) strip.  M multiple of 128.
// Staging uses async global->LDS DMA (no VGPR round trip).
// ---------------------------------------------------------------------------
__global__ void __launch_bounds__(256) gemm_wmma(
    const unsigned short* __restrict__ A, const unsigned short* __restrict__ W,
    const float* __restrict__ bias, void* __restrict__ Cv,
    int N, int Kp, int lda, int ldc, int act, int cF32)
{
  __shared__ __align__(16) unsigned short sA[BM * BK];
  __shared__ __align__(16) unsigned short sW[BN * BK];

  const int tid  = threadIdx.x;
  const int lane = tid & 31;
  const int wave = tid >> 5;
  const int n16  = lane & 15;
  const int h    = lane >> 4;
  const int rowBase = blockIdx.y * BM;
  const int nBase   = blockIdx.x * BN;

  // staging coordinates (hoisted; LDS destinations are loop-invariant)
  const int aRowI = tid >> 1;
  const int aCb   = (tid & 1) * 16;
  const int wRowI = tid >> 2;
  const int wCb   = (tid & 3) * 8;
  const int wN    = nBase + wRowI;
  const unsigned short* wRow = W + (size_t)((wN < N) ? wN : (N - 1)) * Kp;
  const unsigned short* aRow = A + (size_t)(rowBase + aRowI) * lda;
  const unsigned ldsA0 = lds_addr(&sA[aRowI * BK + aCb + 0]);
  const unsigned ldsA1 = lds_addr(&sA[aRowI * BK + aCb + 8]);
  const unsigned ldsW0 = lds_addr(&sW[wRowI * BK + wCb]);

  v8f acc[4] = {};

  for (int k0 = 0; k0 < Kp; k0 += BK) {
    // ---- stage A (BM x BK) and W (BN x BK): async DMA, 16B per op ----
    async_ld_b128(ldsA0, aRow + k0 + aCb + 0);
    async_ld_b128(ldsA1, aRow + k0 + aCb + 8);
    async_ld_b128(ldsW0, wRow + k0 + wCb);
    wait_async0();          // this wave's DMAs landed in LDS
    __syncthreads();        // publish across waves

    // A fragment: rows wave*16 + (lane&15); K runs [8h,8h+8) and [16+8h,24+8h)
    FragBF a;
    const int ar = wave * 16 + n16;
    a.q[0] = *(const uint4*)&sA[ar * BK + 8 * h];
    a.q[1] = *(const uint4*)&sA[ar * BK + 16 + 8 * h];

#pragma unroll
    for (int nt = 0; nt < 4; ++nt) {
      // B fragment: W row nt*16 + (lane&15); K runs [8h,8h+8) and [16+8h,24+8h)
      FragBF bf;
      const int br = nt * 16 + n16;
      bf.q[0] = *(const uint4*)&sW[br * BK + 8 * h];
      bf.q[1] = *(const uint4*)&sW[br * BK + 16 + 8 * h];
      acc[nt] = __builtin_amdgcn_wmma_f32_16x16x32_bf16(
          false, a.v, false, bf.v, (short)0, acc[nt], false, false);
    }
    __syncthreads();
  }

  // ---- epilogue: bias + activation; C/D layout row = r + 8h, col = lane&15
  const int mBase = rowBase + wave * 16;
#pragma unroll
  for (int nt = 0; nt < 4; ++nt) {
    const int col = nBase + nt * 16 + n16;
    if (col < N) {
      const float bv = bias[col];
      size_t idx = (size_t)(mBase + 8 * h) * (size_t)ldc + (size_t)col;
#pragma unroll
      for (int r = 0; r < 8; ++r, idx += (size_t)ldc) {
        float v = acc[nt][r] + bv;
        v = (act == 0) ? fmaxf(v, 0.0f) : (1.0f / (1.0f + expf(-v)));
        if (cF32) ((float*)Cv)[idx] = v;
        else      ((unsigned short*)Cv)[idx] = f2bf(v);
      }
    }
  }
}

// ---------------------------------------------------------------------------
// Embedding bag sum. One wave per (table, bag). Bag b of table t covers
// indices [off[t][b], off[t][b+1]) (last bag runs to NNZ). Writes T row 1+t.
// ---------------------------------------------------------------------------
__global__ void __launch_bounds__(256) embedding_bag_kernel(
    const int* __restrict__ lS_i, const int* __restrict__ lS_o,
    const float* __restrict__ emb_w, float* __restrict__ T)
{
  const int lane = threadIdx.x & 31;
  const int id   = blockIdx.x * 8 + (threadIdx.x >> 5);
  const int t = id >> 12;          // id / 4096
  const int b = id & (B_SZ - 1);   // id % 4096
  if (t >= NTAB) return;

  const int start = lS_o[t * B_SZ + b];
  const int end   = (b == B_SZ - 1) ? NNZ_SZ : lS_o[t * B_SZ + b + 1];

  const float* tbl = emb_w + (size_t)t * VOCAB_SZ * EMB_D;
  float2 acc = make_float2(0.0f, 0.0f);
  for (int j = start; j < end; ++j) {
    const int idx = lS_i[t * NNZ_SZ + j];
    const float2 rv = *(const float2*)(tbl + (size_t)idx * EMB_D + lane * 2);
    acc.x += rv.x;
    acc.y += rv.y;
  }
  float* out = T + (size_t)b * TSTRIDE + (size_t)(t + 1) * EMB_D + lane * 2;
  *(float2*)out = acc;
}

// ---------------------------------------------------------------------------
// Interaction: per batch row, Z = T_b (27x64) * T_b^T via WMMA (padded 32x32).
// Z is parked in LDS (static offsets, no exec juggling), then the 351 strict
// lower-triangle entries are compacted branch-free into R[:,64:415] (bf16).
// Also copies dense x into R[:,0:64] and zeroes pad col 415. 1 wave / batch.
// ---------------------------------------------------------------------------
__global__ void __launch_bounds__(256) interact_kernel(
    const float* __restrict__ T, unsigned short* __restrict__ R)
{
  __shared__ __align__(16) unsigned short sT[8][32 * EMB_D];

  const int lane = threadIdx.x & 31;
  const int wave = threadIdx.x >> 5;
  const int b = blockIdx.x * 8 + wave;
  const float* Tb = T + (size_t)b * TSTRIDE;

  // stage T_b as bf16, rows 27..31 zero-padded. Lane = row; rows 256B aligned.
  {
    const int row = lane;
    const bool valid = (row < TROWS);
    const float* src = Tb + (size_t)(valid ? row : 0) * EMB_D;
#pragma unroll
    for (int c = 0; c < EMB_D; c += 8) {
      float4 f0 = *(const float4*)(src + c);
      float4 f1 = *(const float4*)(src + c + 4);
      if (!valid) {
        f0 = make_float4(0.f, 0.f, 0.f, 0.f);
        f1 = make_float4(0.f, 0.f, 0.f, 0.f);
      }
      *(uint4*)&sT[wave][row * EMB_D + c] =
          make_uint4(pk2(f0.x, f0.y), pk2(f0.z, f0.w), pk2(f1.x, f1.y), pk2(f1.z, f1.w));
    }
  }
  // dense part of R: two bf16 per lane, packed dword store
  unsigned short* rb = R + (size_t)b * RPAD;
  {
    const float2 d = *(const float2*)(Tb + lane * 2);
    *(unsigned int*)(rb + lane * 2) = pk2(d.x, d.y);
  }
  if (lane == 0) rb[RCOLS] = 0;   // zero the stride-pad column

  __syncthreads();

  const int n16 = lane & 15;
  const int h   = lane >> 4;
  v8f acc[2][2] = {};

#pragma unroll
  for (int kk = 0; kk < EMB_D; kk += 32) {
    FragBF a[2], bf[2];
#pragma unroll
    for (int mt = 0; mt < 2; ++mt) {
      const int row = mt * 16 + n16;
      a[mt].q[0] = *(const uint4*)&sT[wave][row * EMB_D + kk + 8 * h];
      a[mt].q[1] = *(const uint4*)&sT[wave][row * EMB_D + kk + 16 + 8 * h];
      bf[mt].q[0] = *(const uint4*)&sT[wave][row * EMB_D + kk + 8 * h];
      bf[mt].q[1] = *(const uint4*)&sT[wave][row * EMB_D + kk + 16 + 8 * h];
    }
#pragma unroll
    for (int mt = 0; mt < 2; ++mt)
#pragma unroll
      for (int nt = 0; nt < 2; ++nt)
        acc[mt][nt] = __builtin_amdgcn_wmma_f32_16x16x32_bf16(
            false, a[mt].v, false, bf[nt].v, (short)0, acc[mt][nt], false, false);
  }

  // park Z (bf16) in this wave's LDS slab -- static offsets, unguarded.
  // sT no longer needed (fragments live in registers; same-wave DS in-order).
  unsigned short* sZ = &sT[wave][0];   // 32x32 bf16 = 2KB of the 4KB slab
#pragma unroll
  for (int mt = 0; mt < 2; ++mt)
#pragma unroll
    for (int nt = 0; nt < 2; ++nt)
#pragma unroll
      for (int r = 0; r < 8; ++r)
        sZ[(mt * 16 + r + 8 * h) * 32 + nt * 16 + n16] = f2bf(acc[mt][nt][r]);

  // branch-free compaction of the 351 strict-lower-triangle entries.
  // f = lane*11+q spans [0,352); clamp 351->350 duplicates one benign store.
  unsigned short* rz = rb + EMB_D;
#pragma unroll
  for (int q = 0; q < 11; ++q) {
    int f = lane * 11 + q;
    f = (f < NTRI) ? f : (NTRI - 1);
    const int i = (int)((1.0f + sqrtf((float)(8 * f + 1))) * 0.5f);
    const int j = f - (i * (i - 1)) / 2;
    rz[f] = sZ[i * 32 + j];
  }
}

// ---------------------------------------------------------------------------
// Launch: converts (weights + dense input) -> bot MLP (last layer writes
// T row 0, f32) -> embedding bags -> interaction -> top MLP (sigmoid, f32).
// ---------------------------------------------------------------------------
extern "C" void kernel_launch(void* const* d_in, const int* in_sizes, int n_in,
                              void* d_out, int out_size, void* d_ws, size_t ws_size,
                              hipStream_t stream) {
  const float* dense_x = (const float*)d_in[0];
  const int*   lS_i    = (const int*)d_in[1];
  const int*   lS_o    = (const int*)d_in[2];
  const float* emb_w   = (const float*)d_in[3];
  const float* bot_w0  = (const float*)d_in[4];
  const float* bot_b0  = (const float*)d_in[5];
  const float* bot_w1  = (const float*)d_in[6];
  const float* bot_b1  = (const float*)d_in[7];
  const float* bot_w2  = (const float*)d_in[8];
  const float* bot_b2  = (const float*)d_in[9];
  const float* top_w0  = (const float*)d_in[10];
  const float* top_b0  = (const float*)d_in[11];
  const float* top_w1  = (const float*)d_in[12];
  const float* top_b1  = (const float*)d_in[13];
  const float* top_w2  = (const float*)d_in[14];
  const float* top_b2  = (const float*)d_in[15];

  // ---- workspace carve (all chunks multiple of 16 bytes) ----
  char* p = (char*)d_ws;
  float* T = (float*)p;              p += (size_t)B_SZ * TSTRIDE * 4;  // f32
  unsigned short* x0b = (unsigned short*)p;  p += (size_t)B_SZ * 32 * 2;
  unsigned short* h0  = (unsigned short*)p;  p += (size_t)B_SZ * 512 * 2;
  unsigned short* h1  = (unsigned short*)p;  p += (size_t)B_SZ * 256 * 2;
  unsigned short* R   = (unsigned short*)p;  p += (size_t)B_SZ * RPAD * 2;
  unsigned short* t0  = (unsigned short*)p;  p += (size_t)B_SZ * 512 * 2;
  unsigned short* t1  = (unsigned short*)p;  p += (size_t)B_SZ * 256 * 2;
  unsigned short* w0b = (unsigned short*)p;  p += (size_t)512 * 32 * 2;
  unsigned short* w1b = (unsigned short*)p;  p += (size_t)256 * 512 * 2;
  unsigned short* w2b = (unsigned short*)p;  p += (size_t)64 * 256 * 2;
  unsigned short* u0b = (unsigned short*)p;  p += (size_t)512 * RPAD * 2;
  unsigned short* u1b = (unsigned short*)p;  p += (size_t)256 * 512 * 2;
  unsigned short* u2b = (unsigned short*)p;  p += (size_t)1 * 256 * 2;

  const dim3 blk(256);
  auto cvt = [&](const float* w, unsigned short* o, int Nrows, int K, int Kp) {
    convert_pad_rows<<<dim3((Kp + 255) / 256, Nrows), blk, 0, stream>>>(w, o, K, Kp);
  };

  // bf16 conversion + zero-padding: weights and dense input
  cvt(dense_x, x0b, B_SZ, 13, 32);
  cvt(bot_w0, w0b, 512, 13, 32);
  cvt(bot_w1, w1b, 256, 512, 512);
  cvt(bot_w2, w2b, 64, 256, 256);
  cvt(top_w0, u0b, 512, RCOLS, RPAD);
  cvt(top_w1, u1b, 256, 512, 512);
  cvt(top_w2, u2b, 1, 256, 256);

  // bottom MLP: 13 -> 512 -> 256 -> 64 (relu), layer 2 writes T[:,0,:] (f32)
  gemm_wmma<<<dim3(512 / BN, B_SZ / BM), blk, 0, stream>>>(
      x0b, w0b, bot_b0, h0, 512, 32, 32, 512, 0, 0);
  gemm_wmma<<<dim3(256 / BN, B_SZ / BM), blk, 0, stream>>>(
      h0, w1b, bot_b1, h1, 256, 512, 512, 256, 0, 0);
  gemm_wmma<<<dim3(64 / BN, B_SZ / BM), blk, 0, stream>>>(
      h1, w2b, bot_b2, T, 64, 256, 256, TSTRIDE, 0, 1);

  // embedding bags -> T[:,1:27,:]
  embedding_bag_kernel<<<dim3(NTAB * B_SZ / 8), blk, 0, stream>>>(lS_i, lS_o, emb_w, T);

  // pairwise interaction -> R (bf16, stride RPAD, col 415 zeroed)
  interact_kernel<<<dim3(B_SZ / 8), blk, 0, stream>>>(T, R);

  // top MLP: 415 -> 512 -> 256 (relu) -> 1 (sigmoid, f32 out)
  gemm_wmma<<<dim3(512 / BN, B_SZ / BM), blk, 0, stream>>>(
      R, u0b, top_b0, t0, 512, RPAD, RPAD, 512, 0, 0);
  gemm_wmma<<<dim3(256 / BN, B_SZ / BM), blk, 0, stream>>>(
      t0, u1b, top_b1, t1, 256, 512, 512, 256, 0, 0);
  gemm_wmma<<<dim3(1, B_SZ / BM), blk, 0, stream>>>(
      t1, u2b, top_b2, d_out, 1, 256, 256, 1, 1, 1);
}
